// MPNN_ENN_Set2Set_22153441313213
// MI455X (gfx1250) — matrix operand order repaired and checked by reading, not compiled, not run on hardware
//
#include <hip/hip_runtime.h>
#include <hip/hip_bf16.h>
#include <math.h>

// ---- problem dims (fixed by the reference) ----
#define N_NODES 8192
#define N_EDGES 32768
#define NB      64
#define NFD     32
#define EFD     16
#define D       64
#define OUTD    12
#define T_MP    3
#define T_S2S   12

typedef __attribute__((ext_vector_type(16))) __bf16 v16bf;
typedef __attribute__((ext_vector_type(8)))  float  v8f;

struct bf16x16_raw { uint4 a, b; };   // 32 bytes = 16 bf16

__device__ __forceinline__ v16bf load_bf16x16(const unsigned short* p) {
    bf16x16_raw r = *(const bf16x16_raw*)p;   // 32B-aligned by construction
    return __builtin_bit_cast(v16bf, r);
}

__device__ __forceinline__ v8f wmma_bf16(v16bf a, v16bf b, v8f c) {
    // (neg_a, A, neg_b, B, c_mod, C, reuse_a, reuse_b)
    return __builtin_amdgcn_wmma_f32_16x16x32_bf16(false, a, false, b, (short)0, c, false, false);
}

__device__ __forceinline__ float sigmoidf_(float x) { return 1.f / (1.f + __expf(-x)); }

// ---- CDNA5 async global->LDS DMA (ASYNCcnt-tracked), per-lane 16B copies ----
__device__ __forceinline__ void async_b128(unsigned lds_byte_addr, const void* gaddr) {
    asm volatile("global_load_async_to_lds_b128 %0, %1, off"
                 :: "v"(lds_byte_addr), "v"(gaddr) : "memory");
}
#define WAIT_ASYNCCNT(N) asm volatile("s_wait_asynccnt %0" :: "n"(N) : "memory")

// ======================================================================
// Small elementwise / packing kernels
// ======================================================================

__global__ void k_zero(float* __restrict__ p, int n) {
    int i = blockIdx.x * blockDim.x + threadIdx.x;
    if (i < n) p[i] = 0.f;
}

// x = node_features @ W_in + b_in   [N,32]@[32,64]
__global__ void k_x0(const float* __restrict__ nf, const float* __restrict__ W,
                     const float* __restrict__ b, float* __restrict__ x) {
    int i = blockIdx.x * blockDim.x + threadIdx.x;
    if (i >= N_NODES * D) return;
    int n = i >> 6, d = i & 63;
    float acc = b[d];
    const float* r = nf + (size_t)n * NFD;
    #pragma unroll
    for (int k = 0; k < NFD; ++k) acc += r[k] * W[k * D + d];
    x[i] = acc;
}

// ed = relu(edge_features @ W_ee1 + b_ee1)   [E,16]@[16,64]  (fp32 out)
__global__ void k_ed(const float* __restrict__ ef, const float* __restrict__ W,
                     const float* __restrict__ b, float* __restrict__ ed) {
    int i = blockIdx.x * blockDim.x + threadIdx.x;
    if (i >= N_EDGES * D) return;
    int e = i >> 6, d = i & 63;
    float acc = b[d];
    const float* r = ef + (size_t)e * EFD;
    #pragma unroll
    for (int k = 0; k < EFD; ++k) acc += r[k] * W[k * D + d];
    ed[i] = fmaxf(acc, 0.f);
}

// W2bf[65][4096] bf16 : rows 0..63 = W_ee2, row 64 = b_ee2 (bias folded as K-slice)
__global__ void k_pack_w2(const float* __restrict__ W_ee2, const float* __restrict__ b_ee2,
                          unsigned short* __restrict__ W2bf) {
    int idx = blockIdx.x * blockDim.x + threadIdx.x;
    if (idx >= 65 * 4096) return;
    int row = idx >> 12, colj = idx & 4095;
    float v = (row < 64) ? W_ee2[idx] : b_ee2[colj];
    W2bf[idx] = __builtin_bit_cast(unsigned short, (__bf16)v);
}

// GRU weights repacked & transposed: Wgt[c][k], c in [0,256): 4 groups of 64
//   g0: s_r = ir+hr, g1: s_z = iz+hz, g2: i_n, g3: h_n ;  k<64 from W_ih, k>=64 from W_hh
__global__ void k_pack_gru(const float* __restrict__ W_ih, const float* __restrict__ W_hh,
                           const float* __restrict__ b_ih, const float* __restrict__ b_hh,
                           unsigned short* __restrict__ Wgt, float* __restrict__ bg) {
    int idx = blockIdx.x * blockDim.x + threadIdx.x;
    if (idx < 256 * 128) {
        int c = idx >> 7, k = idx & 127;
        int grp = c >> 6, d = c & 63;
        float v = 0.f;
        if (k < 64) {
            if (grp == 0) v = W_ih[k * 192 + d];
            else if (grp == 1) v = W_ih[k * 192 + 64 + d];
            else if (grp == 2) v = W_ih[k * 192 + 128 + d];
        } else {
            int kk = k - 64;
            if (grp == 0) v = W_hh[kk * 192 + d];
            else if (grp == 1) v = W_hh[kk * 192 + 64 + d];
            else if (grp == 3) v = W_hh[kk * 192 + 128 + d];
        }
        Wgt[(size_t)c * 128 + k] = __builtin_bit_cast(unsigned short, (__bf16)v);
    } else if (idx < 256 * 128 + 256) {
        int c = idx - 256 * 128;
        int grp = c >> 6, d = c & 63;
        float v;
        if (grp == 0)      v = b_ih[d] + b_hh[d];
        else if (grp == 1) v = b_ih[64 + d] + b_hh[64 + d];
        else if (grp == 2) v = b_ih[128 + d];
        else               v = b_hh[128 + d];
        bg[c] = v;
    }
}

// graph start offsets: gstart[b] = lower_bound(batch, b) (batch is sorted)
__global__ void k_gstart(const int* __restrict__ batch, int* __restrict__ gstart) {
    int b = threadIdx.x;
    if (b > NB) return;
    int lo = 0, hi = N_NODES;
    while (lo < hi) { int mid = (lo + hi) >> 1; if (batch[mid] < b) lo = mid + 1; else hi = mid; }
    gstart[b] = lo;
}

// ======================================================================
// Fused edge-network message GEMM + scatter (the heavy WMMA kernel)
// GEMM:  msg[E,64] = OP[E,4096+64] @ W2bf[4096+64,64]
// where  OP[e, kc*64+j] = ed[e,kc] * h[Esrc[e], j]  (built in registers;
// kc==64 slice has scale 1.0 and carries the b_ee2 bias).
// One wave = 16-edge output tile (16x64), acc in 4x v8f.
// B operand (8KB row of W2bf per kc) is DMA'd into LDS once per block with
// global_load_async_to_lds_b128, double-buffered so the DMA for row kc+1
// overlaps the WMMAs on row kc.
// ======================================================================
__global__ void __launch_bounds__(128)
k_mp_msg(const float* __restrict__ ed, const float* __restrict__ h,
         const int* __restrict__ Esrc, const int* __restrict__ Etgt,
         const unsigned short* __restrict__ W2bf, float* __restrict__ agg) {
    __shared__ unsigned short sbuf[2][4096];   // 2 x 8KB double buffer

    const int tid  = threadIdx.x;
    const int lane = tid & 31;
    const int wave = tid >> 5;
    const int tile = blockIdx.x * (blockDim.x >> 5) + wave;
    const int g    = lane >> 4;        // half-wave group
    const int col  = lane & 15;
    const int mA   = (tile << 4) + col;   // A-fragment row = edge index
    const int src  = Esrc[mA];

    // gather h_src once: 4 runs of 8 (j = 16g + {0,8,32,40} + 0..7), kept fp32
    float hr[4][8];
    const float* hp = h + (size_t)src * D;
    #pragma unroll
    for (int r = 0; r < 4; ++r) {
        int jb = 16 * g + ((r & 1) << 3) + ((r >> 1) << 5);
        float4 p0 = *(const float4*)(hp + jb);
        float4 p1 = *(const float4*)(hp + jb + 4);
        hr[r][0] = p0.x; hr[r][1] = p0.y; hr[r][2] = p0.z; hr[r][3] = p0.w;
        hr[r][4] = p1.x; hr[r][5] = p1.y; hr[r][6] = p1.z; hr[r][7] = p1.w;
    }

    v8f acc[4] = {};
    const float* edrow = ed + (size_t)mA * D;

    // LDS byte addresses of the two staging buffers (per-lane slice = tid*16)
    const unsigned lb0 = (unsigned)(uintptr_t)(&sbuf[0][0]) + (unsigned)tid * 16u;
    const unsigned lb1 = (unsigned)(uintptr_t)(&sbuf[1][0]) + (unsigned)tid * 16u;

    // prologue: DMA row kc=0 into buffer 0 (4 x 2KB issues -> ASYNCcnt=4)
    {
        const char* gp = (const char*)W2bf + (size_t)tid * 16;
        #pragma unroll
        for (int i = 0; i < 4; ++i) async_b128(lb0 + i * 2048u, gp + i * 2048);
    }

    for (int kc = 0; kc <= 64; ++kc) {
        if (kc < 64) {
            // prefetch row kc+1 into the other buffer (its readers finished at
            // the end-of-iteration barrier of kc-1)
            const char* gp = (const char*)(W2bf + (size_t)(kc + 1) * 4096) + (size_t)tid * 16;
            unsigned lb = ((kc + 1) & 1) ? lb1 : lb0;
            #pragma unroll
            for (int i = 0; i < 4; ++i) async_b128(lb + i * 2048u, gp + i * 2048);
            WAIT_ASYNCCNT(4);   // row kc complete; row kc+1 still in flight
        } else {
            WAIT_ASYNCCNT(0);   // last (bias) row complete
        }
        __syncthreads();        // publish DMA'd LDS data to all waves

        const float sf = (kc < 64) ? edrow[kc] : 1.0f;
        const unsigned short* wrow = &sbuf[kc & 1][0];
        #pragma unroll
        for (int half = 0; half < 2; ++half) {
            v16bf afrag;   // A elem e: K = e+16g  -> j runs at 16g+32*half (+8)
            #pragma unroll
            for (int t = 0; t < 8; ++t) {
                afrag[t]     = (__bf16)(sf * hr[2 * half][t]);
                afrag[t + 8] = (__bf16)(sf * hr[2 * half + 1][t]);
            }
            #pragma unroll
            for (int nt = 0; nt < 4; ++nt) {
                const unsigned short* bp = wrow + (nt * 16 + col) * 64 + half * 32 + g * 16;
                acc[nt] = wmma_bf16(afrag, load_bf16x16(bp), acc[nt]);
            }
        }
        __syncthreads();        // all waves done reading before buffer reuse
    }

    // scatter-add into agg[N,64]; C layout: lane holds n=col, rows m = r + 8g
    int tgt[8];
    #pragma unroll
    for (int r = 0; r < 8; ++r) tgt[r] = Etgt[(tile << 4) + 8 * g + r];
    #pragma unroll
    for (int nt = 0; nt < 4; ++nt) {
        #pragma unroll
        for (int r = 0; r < 8; ++r) {
            __hip_atomic_fetch_add(agg + (size_t)tgt[r] * D + nt * 16 + col, acc[nt][r],
                                   __ATOMIC_RELAXED, __HIP_MEMORY_SCOPE_AGENT);
        }
    }
}

// ======================================================================
// GRU node update via WMMA: G[N,256] = [agg|h] (N,128) @ Wgt^T ; gates lane-local
// ======================================================================
__global__ void __launch_bounds__(128)
k_gru(float* __restrict__ h, const float* __restrict__ agg,
      const unsigned short* __restrict__ Wgt, const float* __restrict__ bg) {
    const int lane = threadIdx.x & 31;
    const int wave = threadIdx.x >> 5;
    const int tile = blockIdx.x * (blockDim.x >> 5) + wave;
    const int g    = lane >> 4;
    const int col  = lane & 15;
    const int mA   = (tile << 4) + col;

    // Build 4 A fragments (K=128): j<64 -> agg row, j>=64 -> h row, as bf16
    v16bf afrag[4];
    #pragma unroll
    for (int ks = 0; ks < 4; ++ks) {
        #pragma unroll
        for (int run = 0; run < 2; ++run) {
            int j = 32 * ks + 16 * g + 8 * run;
            const float* sp = (j < 64) ? (agg + (size_t)mA * D + j)
                                       : (h   + (size_t)mA * D + (j - 64));
            float4 p0 = *(const float4*)(sp);
            float4 p1 = *(const float4*)(sp + 4);
            int e0 = run * 8;
            afrag[ks][e0 + 0] = (__bf16)p0.x; afrag[ks][e0 + 1] = (__bf16)p0.y;
            afrag[ks][e0 + 2] = (__bf16)p0.z; afrag[ks][e0 + 3] = (__bf16)p0.w;
            afrag[ks][e0 + 4] = (__bf16)p1.x; afrag[ks][e0 + 5] = (__bf16)p1.y;
            afrag[ks][e0 + 6] = (__bf16)p1.z; afrag[ks][e0 + 7] = (__bf16)p1.w;
        }
    }

    v8f acc[16] = {};
    #pragma unroll
    for (int ks = 0; ks < 4; ++ks) {
        #pragma unroll
        for (int nt = 0; nt < 16; ++nt) {
            const unsigned short* bp = Wgt + (size_t)(nt * 16 + col) * 128 + 32 * ks + 16 * g;
            acc[nt] = wmma_bf16(afrag[ks], load_bf16x16(bp), acc[nt]);
        }
    }

    // gates: cols d, d+64, d+128, d+192 live in the same lane (tiles nt, nt+4, nt+8, nt+12)
    #pragma unroll
    for (int nt = 0; nt < 4; ++nt) {
        int d = nt * 16 + col;
        float b_r = bg[d], b_z = bg[64 + d], b_ni = bg[128 + d], b_nh = bg[192 + d];
        #pragma unroll
        for (int r = 0; r < 8; ++r) {
            int node = (tile << 4) + 8 * g + r;
            float rg = sigmoidf_(acc[nt][r] + b_r);
            float zg = sigmoidf_(acc[nt + 4][r] + b_z);
            float ng = tanhf(acc[nt + 8][r] + b_ni + rg * (acc[nt + 12][r] + b_nh));
            float ho = h[(size_t)node * D + d];
            h[(size_t)node * D + d] = (1.f - zg) * ng + zg * ho;   // in-place: rows private to wave
        }
    }
}

// ======================================================================
// Set2Set
// ======================================================================
__global__ void k_lstm(const float* __restrict__ qstar, float* __restrict__ hS, float* __restrict__ cS,
                       const float* __restrict__ Wl_ih, const float* __restrict__ Wl_hh,
                       const float* __restrict__ bl_ih, const float* __restrict__ bl_hh) {
    int b = blockIdx.x, tid = threadIdx.x;
    __shared__ float qs[128];
    __shared__ float hb[64];
    __shared__ float gates[256];
    if (tid < 128) qs[tid] = qstar[b * 128 + tid];
    if (tid < 64)  hb[tid] = hS[b * 64 + tid];
    __syncthreads();
    {
        int c = tid;
        float acc = bl_ih[c] + bl_hh[c];
        #pragma unroll 4
        for (int k = 0; k < 128; ++k) acc += qs[k] * Wl_ih[k * 256 + c];
        #pragma unroll 4
        for (int k = 0; k < 64; ++k)  acc += hb[k] * Wl_hh[k * 256 + c];
        gates[c] = acc;
    }
    __syncthreads();
    if (tid < 64) {
        float ig = gates[tid], fg = gates[64 + tid], gg = gates[128 + tid], og = gates[192 + tid];
        float cn = sigmoidf_(fg) * cS[b * 64 + tid] + sigmoidf_(ig) * tanhf(gg);
        cS[b * 64 + tid] = cn;
        hS[b * 64 + tid] = sigmoidf_(og) * tanhf(cn);
    }
}

__global__ void k_attn(const float* __restrict__ x, const int* __restrict__ gstart,
                       const float* __restrict__ hS, float* __restrict__ qstar) {
    int b = blockIdx.x, tid = threadIdx.x;
    __shared__ float q[64];
    __shared__ float red[256];
    __shared__ float rpart[4][64];
    __shared__ float ebuf[N_NODES];   // worst case: all nodes in one graph (32 KB)
    int n0 = gstart[b], n1 = gstart[b + 1];
    int cnt = n1 - n0;
    if (tid < 64) q[tid] = hS[b * 64 + tid];
    __syncthreads();

    // phase A: e_n = x[n].q ; block max
    float lmax = -3.4e38f;
    for (int n = tid; n < cnt; n += 256) {
        const float* xr = x + (size_t)(n0 + n) * D;
        float dsum = 0.f;
        #pragma unroll 8
        for (int k = 0; k < D; ++k) dsum += xr[k] * q[k];
        ebuf[n] = dsum;
        lmax = fmaxf(lmax, dsum);
    }
    red[tid] = lmax; __syncthreads();
    for (int s = 128; s > 0; s >>= 1) { if (tid < s) red[tid] = fmaxf(red[tid], red[tid + s]); __syncthreads(); }
    float emax = red[0]; __syncthreads();

    // phase B: exp + denom
    float lsum = 0.f;
    for (int n = tid; n < cnt; n += 256) { float ex = __expf(ebuf[n] - emax); ebuf[n] = ex; lsum += ex; }
    red[tid] = lsum; __syncthreads();
    for (int s = 128; s > 0; s >>= 1) { if (tid < s) red[tid] += red[tid + s]; __syncthreads(); }
    float denom = red[0]; __syncthreads();

    // phase C: r[d] = sum ex_n * x[n,d] / denom   (4 deterministic partial streams)
    int d = tid & 63, ch = tid >> 6;
    float racc = 0.f;
    for (int n = ch; n < cnt; n += 4) racc += ebuf[n] * x[(size_t)(n0 + n) * D + d];
    rpart[ch][d] = racc; __syncthreads();
    if (tid < 64) {
        float rv = (cnt > 0) ? (rpart[0][tid] + rpart[1][tid] + rpart[2][tid] + rpart[3][tid]) / denom : 0.f;
        qstar[b * 128 + tid]      = q[tid];   // q part
        qstar[b * 128 + 64 + tid] = rv;       // r part
    }
}

__global__ void k_out(const float* __restrict__ qstar, const float* __restrict__ W,
                      const float* __restrict__ bo, float* __restrict__ out) {
    int i = threadIdx.x;
    if (i >= NB * OUTD) return;
    int b = i / OUTD, o = i % OUTD;
    float acc = bo[o];
    const float* qr = qstar + (size_t)b * 128;   // first D dims of q_star
    #pragma unroll 8
    for (int d = 0; d < D; ++d) acc += qr[d] * W[d * OUTD + o];
    out[i] = acc;
}

// ======================================================================
// Host-side launch chain
// ======================================================================
static inline char* alignp(char* p, size_t a) {
    return (char*)(((uintptr_t)p + (a - 1)) & ~(uintptr_t)(a - 1));
}

extern "C" void kernel_launch(void* const* d_in, const int* in_sizes, int n_in,
                              void* d_out, int out_size, void* d_ws, size_t ws_size,
                              hipStream_t stream) {
    const float* node_features = (const float*)d_in[0];
    const float* edge_features = (const float*)d_in[1];
    const int*   Esrc  = (const int*)d_in[2];
    const int*   Etgt  = (const int*)d_in[3];
    const int*   batch = (const int*)d_in[4];
    const float* W_in  = (const float*)d_in[5];
    const float* b_in  = (const float*)d_in[6];
    const float* W_ee1 = (const float*)d_in[7];
    const float* b_ee1 = (const float*)d_in[8];
    const float* W_ee2 = (const float*)d_in[9];
    const float* b_ee2 = (const float*)d_in[10];
    const float* W_ih  = (const float*)d_in[11];
    const float* W_hh  = (const float*)d_in[12];
    const float* b_ih  = (const float*)d_in[13];
    const float* b_hh  = (const float*)d_in[14];
    const float* Wl_ih = (const float*)d_in[15];
    const float* Wl_hh = (const float*)d_in[16];
    const float* bl_ih = (const float*)d_in[17];
    const float* bl_hh = (const float*)d_in[18];
    const float* W_out = (const float*)d_in[19];
    const float* b_out = (const float*)d_in[20];
    float* out = (float*)d_out;

    // workspace carve-out
    char* p = (char*)d_ws;
    float* x   = (float*)(p = alignp(p, 256));             p += (size_t)N_NODES * D * 4;   // node state h
    float* ed  = (float*)(p = alignp(p, 256));             p += (size_t)N_EDGES * D * 4;
    float* agg = (float*)(p = alignp(p, 256));             p += (size_t)N_NODES * D * 4;
    unsigned short* W2bf = (unsigned short*)(p = alignp(p, 256)); p += (size_t)65 * 4096 * 2;
    unsigned short* Wgt  = (unsigned short*)(p = alignp(p, 256)); p += (size_t)256 * 128 * 2;
    float* bg    = (float*)(p = alignp(p, 256));           p += 256 * 4;
    float* state = (float*)(p = alignp(p, 256));           p += (size_t)NB * 256 * 4;      // hS|cS|qstar
    int*   gstart= (int*)(p = alignp(p, 256));             p += (NB + 1) * 4;
    float* hS = state;
    float* cS = state + NB * 64;
    float* qstar = state + NB * 128;

    // ---- one-time preprocessing ----
    k_x0<<<(N_NODES * D) / 256, 256, 0, stream>>>(node_features, W_in, b_in, x);
    k_ed<<<(N_EDGES * D) / 256, 256, 0, stream>>>(edge_features, W_ee1, b_ee1, ed);
    k_pack_w2<<<(65 * 4096 + 255) / 256, 256, 0, stream>>>(W_ee2, b_ee2, W2bf);
    k_pack_gru<<<(256 * 128 + 256 + 255) / 256, 256, 0, stream>>>(W_ih, W_hh, b_ih, b_hh, Wgt, bg);
    k_gstart<<<1, 128, 0, stream>>>(batch, gstart);

    // ---- T_MP message-passing steps (WMMA) ----
    for (int t = 0; t < T_MP; ++t) {
        k_zero<<<(N_NODES * D) / 256, 256, 0, stream>>>(agg, N_NODES * D);
        k_mp_msg<<<N_EDGES / 64, 128, 0, stream>>>(ed, x, Esrc, Etgt, W2bf, agg);
        k_gru<<<N_NODES / 64, 128, 0, stream>>>(x, agg, Wgt, bg);
    }

    // ---- Set2Set ----
    k_zero<<<(NB * 256) / 256, 256, 0, stream>>>(state, NB * 256);
    for (int s = 0; s < T_S2S; ++s) {
        k_lstm<<<NB, 256, 0, stream>>>(qstar, hS, cS, Wl_ih, Wl_hh, bl_ih, bl_hh);
        k_attn<<<NB, 256, 0, stream>>>(x, gstart, hS, qstar);
    }

    // ---- output head ----
    k_out<<<1, NB * OUTD, 0, stream>>>(qstar, W_out, b_out, out);
}
